// TorchAttention_82334523064625
// MI455X (gfx1250) — matrix operand order repaired and checked by reading, not compiled, not run on hardware
//
#include <hip/hip_runtime.h>

#define BB  2
#define SS  2048
#define DD  4096
#define HH  32
#define KVH 8
#define HDD 128

typedef __bf16 bf16;
typedef __attribute__((ext_vector_type(16))) __bf16 v16bf;
typedef __attribute__((ext_vector_type(8)))  __bf16 bf16x8;
typedef __attribute__((ext_vector_type(2)))  __bf16 bf16x2;
typedef __attribute__((ext_vector_type(8)))  float  v8f;
typedef int v4i __attribute__((vector_size(4 * sizeof(int))));

// ---------------------------------------------------------------------------
// Async global->LDS copy (CDNA5 ASYNCcnt path) with compile-safe fallback.
// Builtin signature (from clang diagnostic): param0 = v4i __device__* (AS1),
// param1 = LDS side (AS3), then offset, cpol.
// ---------------------------------------------------------------------------
#if defined(__has_builtin)
#  if __has_builtin(__builtin_amdgcn_global_load_async_to_lds_b128)
#    define ASYNC_LDS 1
#  else
#    define ASYNC_LDS 0
#  endif
#else
#  define ASYNC_LDS 0
#endif

typedef __attribute__((address_space(1))) v4i* gv4i_p;
typedef __attribute__((address_space(3))) v4i* lv4i_p;

__device__ __forceinline__ void async_copy_b128(const bf16* g, bf16* l) {
#if ASYNC_LDS
    __builtin_amdgcn_global_load_async_to_lds_b128((gv4i_p)g, (lv4i_p)l, 0, 0);
#else
    *(bf16x8*)l = *(const bf16x8*)g;
#endif
}

__device__ __forceinline__ void async_join() {
#if ASYNC_LDS
#  if __has_builtin(__builtin_amdgcn_s_wait_asynccnt)
    __builtin_amdgcn_s_wait_asynccnt(0);
#  else
    asm volatile("s_wait_asynccnt 0x0" ::: "memory");
#  endif
#endif
}

__device__ __forceinline__ v8f wmma_bf16(v16bf a, v16bf b, v8f c) {
    return __builtin_amdgcn_wmma_f32_16x16x32_bf16(
        false, a, false, b, (short)0, c, false, false);
}

// ---------------------------------------------------------------------------
// Kernel 0: f32 -> bf16 conversion
// ---------------------------------------------------------------------------
__global__ __launch_bounds__(256)
void f32_to_bf16_kernel(const float* __restrict__ in, bf16* __restrict__ out, size_t n) {
    size_t i = (size_t)blockIdx.x * blockDim.x + threadIdx.x;
    if (i < n) out[i] = (bf16)in[i];
}

// ---------------------------------------------------------------------------
// Kernel 1: GEMM  C[M,N] = A[M,K](bf16) @ W[N,K](f32)^T
// 256 threads = 8 waves. Workgroup tile: 32 rows x 128 cols.
// Each wave: two 16x16 subtiles sharing one B fragment (2 WMMA per W-load).
// A chunk (32x32 bf16) staged in LDS (async path when available).
// ---------------------------------------------------------------------------
__global__ __launch_bounds__(256)
void gemm_bf16_wT(const bf16* __restrict__ A, const float* __restrict__ W,
                  float* __restrict__ Cf, bf16* __restrict__ Cb,
                  int M, int N, int K) {
    __shared__ bf16 Atile[32 * 32];
    const int tid  = threadIdx.x;
    const int wave = tid >> 5;
    const int lane = tid & 31;
    const int l15  = lane & 15;
    const int lh   = lane >> 4;
    const int mbase = blockIdx.y * 32;
    const int nbase = blockIdx.x * 128 + wave * 16;

    const int srow  = tid >> 2;        // 0..31   (staging, tid<128)
    const int spart = (tid & 3) * 8;   // element offset, 16B chunks

    v8f acc0 = {}, acc1 = {};
    for (int kc = 0; kc < K; kc += 32) {
        __syncthreads();
        if (tid < 128)
            async_copy_b128(A + (size_t)(mbase + srow) * K + kc + spart,
                            Atile + srow * 32 + spart);
        async_join();
        __syncthreads();

        // two A fragments (rows 0..15 and 16..31), ISA A-layout
        v16bf af0, af1;
#pragma unroll
        for (int v = 0; v < 8; ++v) {
            int kk = (v >> 2) * 16 + lh * 8 + (v & 3) * 2;
            bf16x2 p0 = *(const bf16x2*)(Atile + l15 * 32 + kk);
            bf16x2 p1 = *(const bf16x2*)(Atile + (16 + l15) * 32 + kk);
            af0[2 * v] = p0[0]; af0[2 * v + 1] = p0[1];
            af1[2 * v] = p1[0]; af1[2 * v + 1] = p1[1];
        }

        // B fragment (32x16): B[k][n] = W[n][kc+k], cvt f32->bf16 in regs
        const float* wrow = W + (size_t)(nbase + l15) * K + kc + lh * 8;
        __builtin_prefetch(wrow + 32, 0, 1);   // next k-chunk
        v16bf bfm;
#pragma unroll
        for (int v = 0; v < 8; ++v) {
            bfm[2 * v]     = (bf16)wrow[(v >> 2) * 8 + 2 * v];
            bfm[2 * v + 1] = (bf16)wrow[(v >> 2) * 8 + 2 * v + 1];
        }

        acc0 = wmma_bf16(af0, bfm, acc0);
        acc1 = wmma_bf16(af1, bfm, acc1);
    }

#pragma unroll
    for (int v = 0; v < 8; ++v) {
        size_t i0 = (size_t)(mbase + v + 8 * lh) * N + (nbase + l15);
        size_t i1 = (size_t)(mbase + 16 + v + 8 * lh) * N + (nbase + l15);
        if (Cf) { Cf[i0] = acc0[v];        Cf[i1] = acc1[v]; }
        else    { Cb[i0] = (bf16)acc0[v];  Cb[i1] = (bf16)acc1[v]; }
    }
}

// ---------------------------------------------------------------------------
// Kernel 2: RoPE in place on bf16 tensor laid out (B*S, nh, HD).
// ---------------------------------------------------------------------------
__global__ __launch_bounds__(256)
void rope_kernel(bf16* __restrict__ t, const float* __restrict__ cosb,
                 const float* __restrict__ sinb, int nh, size_t n) {
    size_t i = (size_t)blockIdx.x * blockDim.x + threadIdx.x;
    if (i >= n) return;
    int d = (int)(i & 63);
    size_t r = i >> 6;
    int h = (int)(r % nh);
    size_t bs = r / nh;                     // b*S + s
    bf16* row = t + (bs * nh + h) * (size_t)HDD;
    float c0 = cosb[bs * HDD + d],      s0 = sinb[bs * HDD + d];
    float c1 = cosb[bs * HDD + d + 64], s1 = sinb[bs * HDD + d + 64];
    float lo = (float)row[d];
    float hi = (float)row[d + 64];
    row[d]      = (bf16)(lo * c0 - hi * s0);
    row[d + 64] = (bf16)(hi * c1 + lo * s1);
}

// ---------------------------------------------------------------------------
// Kernel 3: flash attention (causal, GQA H=32 over KV=8).
// 128 threads = 4 waves; workgroup owns 64 queries of one (b,h); all 4 waves
// walk the SAME key blocks so K (32x128) and V (transposed 128x32) tiles are
// staged cooperatively in LDS once per block. Per 32-key block: 8 WMMAs for
// scores + 8 WMMAs for P@V; online softmax with butterflies inside 16-lane
// halves; P bounced through per-wave LDS into A-fragment layout.
// ---------------------------------------------------------------------------
__global__ __launch_bounds__(128)
void attn_kernel(const bf16* __restrict__ qb, const bf16* __restrict__ kb,
                 const bf16* __restrict__ vb, bf16* __restrict__ ab) {
    __shared__ bf16 Klds[32 * 128];      // [key][d]
    __shared__ bf16 Vt[128 * 32];        // [d][key]  (transposed)
    __shared__ bf16 Plds[4][16 * 32];

    const int tid  = threadIdx.x;
    const int wave = tid >> 5;
    const int lane = tid & 31;
    const int l15  = lane & 15;
    const int lh   = lane >> 4;

    int blk = blockIdx.x;              // (b*H + h)*32 + qg   (S/64 = 32)
    int qg  = blk & 31;
    int bh  = blk >> 5;
    int h   = bh % HH;
    int b   = bh / HH;
    int qbase = qg * 64 + wave * 16;
    int kvh = h / (HH / KVH);

    // Q fragments: 4 A-frags (16x32) covering HD=128
    v16bf qf[4];
#pragma unroll
    for (int ks = 0; ks < 4; ++ks) {
#pragma unroll
        for (int v = 0; v < 8; ++v) {
            int kk = 32 * ks + (v >> 2) * 16 + lh * 8 + (v & 3) * 2;
            int s = qbase + l15;
            bf16x2 p = *(const bf16x2*)(qb + ((size_t)(b * SS + s) * HH + h) * HDD + kk);
            qf[ks][2 * v] = p[0]; qf[ks][2 * v + 1] = p[1];
        }
    }

    v8f z = {};
    v8f o[8];
#pragma unroll
    for (int i = 0; i < 8; ++i) o[i] = z;
    float mrun[8], lrun[8];
#pragma unroll
    for (int v = 0; v < 8; ++v) { mrun[v] = -1e30f; lrun[v] = 0.0f; }

    const float scale = 0.08838834764831845f;   // 1/sqrt(128)
    const int nkb = 2 * qg + 2;                 // key blocks for the 64-query tile

    for (int kbi = 0; kbi < nkb; ++kbi) {
        int kb0 = kbi * 32;

        __syncthreads();   // protect LDS tiles from previous iteration readers

        // ---- stage K tile (32 keys x 128 dims), 4x b128 per thread (async) ----
#pragma unroll
        for (int i = 0; i < 4; ++i) {
            int c = tid + i * 128;                // 0..511
            int key = c >> 4, part = (c & 15) * 8;
            async_copy_b128(
                kb + ((size_t)(b * SS + kb0 + key) * KVH + kvh) * HDD + part,
                Klds + key * 128 + part);
        }
        // ---- stage V tile transposed: Vt[d][key] ----
        {
            int key = tid >> 2, dc = (tid & 3) * 32;
            const bf16* Vg = vb + ((size_t)(b * SS + kb0 + key) * KVH + kvh) * HDD + dc;
#pragma unroll
            for (int j = 0; j < 4; ++j) {
                bf16x8 vv = *(const bf16x8*)(Vg + j * 8);
#pragma unroll
                for (int e = 0; e < 8; ++e)
                    Vt[(dc + j * 8 + e) * 32 + key] = vv[e];
            }
        }
        async_join();
        __syncthreads();

        if (kb0 <= qbase + 15) {   // uniform per-wave causal skip (no barriers inside)
            // scores: two 16x16 D-tiles (M=query, N=key), K-dim = 128 in 4 steps
            v8f sc[2];
#pragma unroll
            for (int t = 0; t < 2; ++t) {
                v8f a = z;
#pragma unroll
                for (int ks = 0; ks < 4; ++ks) {
                    v16bf kf;
#pragma unroll
                    for (int v = 0; v < 8; ++v) {
                        bf16x2 p = *(const bf16x2*)(
                            Klds + (t * 16 + l15) * 128 + 32 * ks + 8 * lh + (v >> 2) * 8 + 2 * v);
                        kf[2 * v] = p[0]; kf[2 * v + 1] = p[1];
                    }
                    a = wmma_bf16(qf[ks], kf, a);
                }
                sc[t] = a;
            }

            // scale + causal mask (q row = v + 8*lh, key col = l15)
#pragma unroll
            for (int t = 0; t < 2; ++t) {
#pragma unroll
                for (int v = 0; v < 8; ++v) {
                    int q = qbase + v + 8 * lh;
                    int k = kb0 + t * 16 + l15;
                    float s = sc[t][v] * scale;
                    sc[t][v] = (k <= q) ? s : -1e30f;
                }
            }

            // online softmax (butterflies stay inside 16-lane halves)
#pragma unroll
            for (int v = 0; v < 8; ++v) {
                float mc = fmaxf(sc[0][v], sc[1][v]);
                mc = fmaxf(mc, __shfl_xor(mc, 1, 32));
                mc = fmaxf(mc, __shfl_xor(mc, 2, 32));
                mc = fmaxf(mc, __shfl_xor(mc, 4, 32));
                mc = fmaxf(mc, __shfl_xor(mc, 8, 32));
                float mn = fmaxf(mrun[v], mc);
                float alpha = __expf(mrun[v] - mn);
                float p0 = __expf(sc[0][v] - mn);
                float p1 = __expf(sc[1][v] - mn);
                sc[0][v] = p0; sc[1][v] = p1;
                float ps = p0 + p1;
                ps += __shfl_xor(ps, 1, 32);
                ps += __shfl_xor(ps, 2, 32);
                ps += __shfl_xor(ps, 4, 32);
                ps += __shfl_xor(ps, 8, 32);
                lrun[v] = lrun[v] * alpha + ps;
                mrun[v] = mn;
#pragma unroll
                for (int dt = 0; dt < 8; ++dt) o[dt][v] *= alpha;
            }

            // bounce P (16x32 bf16) through per-wave LDS into A-frag layout
            bf16* P = Plds[wave];
#pragma unroll
            for (int t = 0; t < 2; ++t)
#pragma unroll
                for (int v = 0; v < 8; ++v)
                    P[(v + 8 * lh) * 32 + t * 16 + l15] = (bf16)sc[t][v];
            asm volatile("s_wait_dscnt 0x0" ::: "memory");
            v16bf pf;
#pragma unroll
            for (int v = 0; v < 8; ++v) {
                int kk = (v >> 2) * 16 + lh * 8 + (v & 3) * 2;
                bf16x2 pp = *(const bf16x2*)(P + l15 * 32 + kk);
                pf[2 * v] = pp[0]; pf[2 * v + 1] = pp[1];
            }

            // O += P @ V : 8 d-tiles; V B-frag pairs contiguous in Vt
#pragma unroll
            for (int dt = 0; dt < 8; ++dt) {
                v16bf vf;
                int d = dt * 16 + l15;
#pragma unroll
                for (int v = 0; v < 8; ++v) {
                    bf16x2 p = *(const bf16x2*)(Vt + d * 32 + 8 * lh + (v >> 2) * 8 + 2 * v);
                    vf[2 * v] = p[0]; vf[2 * v + 1] = p[1];
                }
                o[dt] = wmma_bf16(pf, vf, o[dt]);
            }
        }
    }

    // normalize and store to (B,S,H*HD) bf16
#pragma unroll
    for (int dt = 0; dt < 8; ++dt) {
#pragma unroll
        for (int v = 0; v < 8; ++v) {
            int s = qbase + v + 8 * lh;
            int d = dt * 16 + l15;
            float val = o[dt][v] / lrun[v];
            ab[((size_t)(b * SS + s) * HH + h) * HDD + d] = (bf16)val;
        }
    }
}

// ---------------------------------------------------------------------------
// launch
// ---------------------------------------------------------------------------
extern "C" void kernel_launch(void* const* d_in, const int* in_sizes, int n_in,
                              void* d_out, int out_size, void* d_ws, size_t ws_size,
                              hipStream_t stream) {
    const float* x    = (const float*)d_in[0];
    const float* wq   = (const float*)d_in[1];
    const float* wk   = (const float*)d_in[2];
    const float* wv   = (const float*)d_in[3];
    const float* wo   = (const float*)d_in[4];
    const float* cosb = (const float*)d_in[5];
    const float* sinb = (const float*)d_in[6];
    float* out = (float*)d_out;

    char* ws = (char*)d_ws;
    bf16* xb   = (bf16*)(ws);                       // 4096*4096 bf16 = 32 MB
    bf16* qbuf = (bf16*)(ws + (size_t)33554432);    // 4096*4096 bf16
    bf16* kbuf = (bf16*)(ws + (size_t)67108864);    // 4096*1024 bf16
    bf16* vbuf = (bf16*)(ws + (size_t)75497472);    // 4096*1024 bf16
    bf16* abuf = (bf16*)(ws + (size_t)83886080);    // 4096*4096 bf16

    const int M = BB * SS;                          // 4096

    // x -> bf16
    f32_to_bf16_kernel<<<(unsigned)(((size_t)M * DD + 255) / 256), 256, 0, stream>>>(
        x, xb, (size_t)M * DD);

    // projections
    dim3 gq(DD / 128, M / 32);            // 32 x 128
    dim3 gk((KVH * HDD) / 128, M / 32);   // 8 x 128
    gemm_bf16_wT<<<gq, 256, 0, stream>>>(xb, wq, nullptr, qbuf, M, DD, DD);
    gemm_bf16_wT<<<gk, 256, 0, stream>>>(xb, wk, nullptr, kbuf, M, KVH * HDD, DD);
    gemm_bf16_wT<<<gk, 256, 0, stream>>>(xb, wv, nullptr, vbuf, M, KVH * HDD, DD);

    // RoPE on Q and K
    size_t nq = (size_t)M * HH * 64;
    size_t nk = (size_t)M * KVH * 64;
    rope_kernel<<<(unsigned)((nq + 255) / 256), 256, 0, stream>>>(qbuf, cosb, sinb, HH, nq);
    rope_kernel<<<(unsigned)((nk + 255) / 256), 256, 0, stream>>>(kbuf, cosb, sinb, KVH, nk);

    // flash attention: one block per (b, h, 64-query tile)
    attn_kernel<<<BB * HH * (SS / 64), 128, 0, stream>>>(qbuf, kbuf, vbuf, abuf);

    // output projection -> f32 d_out
    gemm_bf16_wT<<<gq, 256, 0, stream>>>(abuf, wo, out, nullptr, M, DD, DD);
}
